// GPMatchEncoderTRT_45732811768429
// MI455X (gfx1250) — matrix operand order repaired and checked by reading, not compiled, not run on hardware
//
#include <hip/hip_runtime.h>
#include <math.h>

typedef _Float16 v8h  __attribute__((ext_vector_type(8)));
typedef _Float16 v16h __attribute__((ext_vector_type(16)));
typedef float    v8f  __attribute__((ext_vector_type(8)));

#define BATCH 4
#define CCH   256
#define NPIX  4096   // 64*64
#define LOG2E 1.4426950408889634f
#define BETA  10.0f

__device__ __forceinline__ float fast_exp2(float x) {
#if __has_builtin(__builtin_amdgcn_exp2f)
    return __builtin_amdgcn_exp2f(x);
#else
    return exp2f(x);
#endif
}

// ---------------------------------------------------------------------------
// Phase 1: L2-normalize each (b, n) row over the (strided) channel dim and
// pack to f16 row-major [b, n, c] in workspace.
// ---------------------------------------------------------------------------
__global__ void norm_pack_kernel(const float* __restrict__ f0,
                                 const float* __restrict__ f1,
                                 _Float16* __restrict__ a_h,
                                 _Float16* __restrict__ b_h) {
    int gid   = blockIdx.x * blockDim.x + threadIdx.x;    // 0 .. 2*B*N-1
    int which = (gid >= BATCH * NPIX) ? 1 : 0;
    int idx   = which ? gid - BATCH * NPIX : gid;
    int b     = idx >> 12;          // / 4096
    int n     = idx & (NPIX - 1);

    const float* src = (which ? f1 : f0) + ((size_t)b * CCH) * NPIX + n;
    _Float16*    dst = (which ? b_h : a_h) + ((size_t)b * NPIX + n) * CCH;

    float ss = 0.0f;
    for (int c = 0; c < CCH; ++c) {
        float v = src[(size_t)c * NPIX];
        ss = fmaf(v, v, ss);
    }
    float rn = 1.0f / fmaxf(sqrtf(ss), 1e-6f);

    for (int c = 0; c < CCH; c += 8) {
        v8h pk;
        #pragma unroll
        for (int j = 0; j < 8; ++j)
            pk[j] = (_Float16)(src[(size_t)(c + j) * NPIX] * rn);
        *(v8h*)(dst + c) = pk;
    }
}

// ---------------------------------------------------------------------------
// Phase 2: fused  sim = a.b^T * beta  -> streamed softmax ->
//          (sum p*x, sum p*y, max p) per query row.
// Each wave: 2 A-tiles (32 query rows), loops over 256 m-tiles of 16 columns,
// 8 K-chunks of 32 per tile -> 16 v_wmma_f32_16x16x32_f16 per m-tile.
// Block = 4 waves = 128 query rows of one batch; grid = 128 blocks.
// ---------------------------------------------------------------------------
__global__ __launch_bounds__(128)
void match_kernel(const _Float16* __restrict__ a_h,
                  const _Float16* __restrict__ b_h,
                  float* __restrict__ out) {
    const int lane = threadIdx.x & 31;
    const int wave = threadIdx.x >> 5;
    const int lmod = lane & 15;
    const int hi   = lane >> 4;                 // 0: lanes 0-15, 1: lanes 16-31

    const int b    = blockIdx.x >> 5;           // 32 blocks per batch
    const int row0 = ((blockIdx.x & 31) << 7) + (wave << 5);

    const _Float16* aB = a_h + ((size_t)b * NPIX) * CCH;
    const _Float16* bB = b_h + ((size_t)b * NPIX) * CCH;

    // ---- load both A tiles (16 rows x K=256) into registers, contiguous k order (same as B) ----
    // lane L<16 row L: elems 0-15 = K 0..15 (per 32-chunk)
    // lane L>=16 row L-16: elems 0-15 = K 16..31
    v16h aT[2][8];
    #pragma unroll
    for (int t = 0; t < 2; ++t) {
        const _Float16* pa = aB + (size_t)(row0 + t * 16 + lmod) * CCH + hi * 16;
        #pragma unroll
        for (int kc = 0; kc < 8; ++kc) {
            v8h lo = *(const v8h*)(pa + kc * 32);
            v8h hh = *(const v8h*)(pa + kc * 32 + 8);
            v16h av;
            #pragma unroll
            for (int j = 0; j < 8; ++j) { av[j] = lo[j]; av[8 + j] = hh[j]; }
            aT[t][kc] = av;
        }
    }

    const float SC = BETA * LOG2E;              // log2 domain, stabilizer = BETA
    float vmax[2][8], s[2][8], sx[2][8], sy[2][8];
    #pragma unroll
    for (int t = 0; t < 2; ++t)
        #pragma unroll
        for (int r = 0; r < 8; ++r) {
            vmax[t][r] = -1e30f;
            s[t][r] = 0.0f; sx[t][r] = 0.0f; sy[t][r] = 0.0f;
        }

    // B operand: lane holds column m = m0+lmod, 16 contiguous K values
    const _Float16* pb0 = bB + (size_t)lmod * CCH + hi * 16;

    for (int m0 = 0; m0 < NPIX; m0 += 16) {
        const _Float16* pb = pb0 + (size_t)m0 * CCH;
        __builtin_prefetch(pb + 16 * CCH, 0, 3);   // next m-tile -> global_prefetch

        v8f acc0 = {}, acc1 = {};
        #pragma unroll
        for (int kc = 0; kc < 8; ++kc) {
            v16h bm = *(const v16h*)(pb + kc * 32);
            acc0 = __builtin_amdgcn_wmma_f32_16x16x32_f16(
                false, aT[0][kc], false, bm, (short)0, acc0, false, false);
            acc1 = __builtin_amdgcn_wmma_f32_16x16x32_f16(
                false, aT[1][kc], false, bm, (short)0, acc1, false, false);
        }

        // streamed softmax update; element r of lane = sim[r + 8*hi][lmod]
        const int   m  = m0 + lmod;
        const float xf = (float)(m & 63);
        const float yf = (float)(m >> 6);
        #pragma unroll
        for (int r = 0; r < 8; ++r) {
            float t0 = fmaf(acc0[r], SC, -SC);      // beta*log2e*(cos-1)
            float p0 = fast_exp2(t0);
            vmax[0][r] = fmaxf(vmax[0][r], t0);
            s [0][r] += p0;
            sx[0][r] = fmaf(p0, xf, sx[0][r]);
            sy[0][r] = fmaf(p0, yf, sy[0][r]);

            float t1 = fmaf(acc1[r], SC, -SC);
            float p1 = fast_exp2(t1);
            vmax[1][r] = fmaxf(vmax[1][r], t1);
            s [1][r] += p1;
            sx[1][r] = fmaf(p1, xf, sx[1][r]);
            sy[1][r] = fmaf(p1, yf, sy[1][r]);
        }
    }

    // ---- XOR butterfly over the 16 lanes of each half (masks stay in-group) ----
    #pragma unroll
    for (int t = 0; t < 2; ++t)
        #pragma unroll
        for (int r = 0; r < 8; ++r)
            #pragma unroll
            for (int mk = 1; mk <= 8; mk <<= 1) {
                vmax[t][r] = fmaxf(vmax[t][r], __shfl_xor(vmax[t][r], mk, 32));
                s [t][r] += __shfl_xor(s [t][r], mk, 32);
                sx[t][r] += __shfl_xor(sx[t][r], mk, 32);
                sy[t][r] += __shfl_xor(sy[t][r], mk, 32);
            }

    // lanes 0 (rows 0-7 of each tile) and 16 (rows 8-15) write results
    if (lmod == 0) {
        float* warp_o = out;                              // [B, N, 2]
        float* cert_o = out + (size_t)BATCH * NPIX * 2;   // [B, N]
        #pragma unroll
        for (int t = 0; t < 2; ++t) {
            int rb = row0 + t * 16 + hi * 8;
            #pragma unroll
            for (int r = 0; r < 8; ++r) {
                int    n   = rb + r;
                float  inv = 1.0f / s[t][r];
                size_t o   = ((size_t)b * NPIX + n) * 2;
                warp_o[o]     = sx[t][r] * inv;
                warp_o[o + 1] = sy[t][r] * inv;
                cert_o[(size_t)b * NPIX + n] = fast_exp2(vmax[t][r]) * inv;
            }
        }
    }
}

// ---------------------------------------------------------------------------
extern "C" void kernel_launch(void* const* d_in, const int* in_sizes, int n_in,
                              void* d_out, int out_size, void* d_ws, size_t ws_size,
                              hipStream_t stream) {
    const float* f0 = (const float*)d_in[0];
    const float* f1 = (const float*)d_in[1];

    _Float16* a_h = (_Float16*)d_ws;                       // 8 MB
    _Float16* b_h = a_h + (size_t)BATCH * NPIX * CCH;      // 8 MB

    // 2 * 4 * 4096 rows, one thread each
    norm_pack_kernel<<<(2 * BATCH * NPIX) / 256, 256, 0, stream>>>(f0, f1, a_h, b_h);

    // 128 blocks x 4 waves; each wave covers 32 query rows
    match_kernel<<<128, 128, 0, stream>>>(a_h, b_h, (float*)d_out);
}
